// FixLenLemmaModel_53420803228466
// MI455X (gfx1250) — matrix operand order repaired and checked by reading, not compiled, not run on hardware
//
#include <hip/hip_runtime.h>
#include <hip/hip_bf16.h>

// ---------------- problem constants (from reference) ----------------
static constexpr int BATCH   = 32;
static constexpr int SEQ     = 2048;
static constexpr int NTOK    = BATCH * SEQ;    // 65536 flat positions
static constexpr int DMODEL  = 768;
static constexpr int NVOCAB  = 1024;           // output vocab
static constexpr int NGROUP  = 8192;
static constexpr int MAXLEN  = 8;
static constexpr int NSLOT   = NGROUP * MAXLEN; // 65536 GEMM rows
static constexpr int KSTEPS  = DMODEL / 32;     // 24

// ---------------- vector types ----------------
typedef __attribute__((ext_vector_type(16))) _Float16 v16h;
typedef __attribute__((ext_vector_type(8)))  _Float16 v8h;
typedef __attribute__((ext_vector_type(8)))  float    v8f;
typedef __attribute__((ext_vector_type(4)))  int      v4i;

// CDNA5 async global->LDS path (guarded; falls back to sync staging)
#if __has_builtin(__builtin_amdgcn_global_load_async_to_lds_b128)
#define FL_ASYNC 1
#else
#define FL_ASYNC 0
#endif

// b128 async copy moves one int32x4 per lane; builtin takes AS1/AS3 v4i*.
#define FL_TO_GLOBAL(p) ((__attribute__((address_space(1))) v4i*)(uintptr_t)(p))
#define FL_TO_LDS(p)    ((__attribute__((address_space(3))) v4i*)(uintptr_t)(p))

__device__ __forceinline__ void fl_wait_async() {
#if FL_ASYNC
#if __has_builtin(__builtin_amdgcn_s_wait_asynccnt)
    __builtin_amdgcn_s_wait_asynccnt(0);
#else
    asm volatile("s_wait_asynccnt 0" ::: "memory");
#endif
#endif
}

// masked f32x8 -> f16x8 convert (mask 0/1 keeps empty slots at zero)
__device__ __forceinline__ v8h fl_cvt8(const float4 f0, const float4 f1, float m) {
    v8h ah;
    ah[0] = (_Float16)(f0.x * m); ah[1] = (_Float16)(f0.y * m);
    ah[2] = (_Float16)(f0.z * m); ah[3] = (_Float16)(f0.w * m);
    ah[4] = (_Float16)(f1.x * m); ah[5] = (_Float16)(f1.y * m);
    ah[6] = (_Float16)(f1.z * m); ah[7] = (_Float16)(f1.w * m);
    return ah;
}

// ==================================================================
// Prep kernel 0: init counts to 0 and slot->token map to -1
// ==================================================================
__global__ void fl_init_kernel(int* __restrict__ cnt, int* __restrict__ amap) {
    const int i = blockIdx.x * blockDim.x + threadIdx.x;
    if (i < NGROUP * BATCH) {
        cnt[i] = 0;
    } else if (i < NGROUP * BATCH + NSLOT) {
        amap[i - NGROUP * BATCH] = -1;
    }
}

// ==================================================================
// Prep kernel 1: per-(group,row) occurrence counts
// ==================================================================
__global__ void fl_count_kernel(const int* __restrict__ target, int* __restrict__ cnt) {
    const int n = blockIdx.x * blockDim.x + threadIdx.x;
    if (n >= NTOK) return;
    const int g   = target[n];
    const int row = n >> 11;              // n / SEQ
    atomicAdd(&cnt[g * BATCH + row], 1);
}

// ==================================================================
// Prep kernel 2: exclusive prefix over the 32 rows of each group
// ==================================================================
__global__ void fl_scan_kernel(int* __restrict__ cnt) {
    const int g = blockIdx.x * blockDim.x + threadIdx.x;
    if (g >= NGROUP) return;
    int sum = 0;
    #pragma unroll
    for (int r = 0; r < BATCH; ++r) {
        const int c = cnt[g * BATCH + r];
        cnt[g * BATCH + r] = sum;
        sum += c;
    }
}

// ==================================================================
// Prep kernel 3: rank of each position inside its group.
// target sorted within each row -> within-row rank = s - lower_bound.
// Global rank = per-row base + within-row rank; rank>=MAXLEN truncated.
// ==================================================================
__global__ void fl_rank_kernel(const int* __restrict__ inputs,
                               const int* __restrict__ target,
                               const int* __restrict__ off,
                               int* __restrict__ amap) {
    const int n = blockIdx.x * blockDim.x + threadIdx.x;
    if (n >= NTOK) return;
    const int row = n >> 11;
    const int s   = n & (SEQ - 1);
    const int g   = target[n];
    const int* trow = target + row * SEQ;
    int lo = 0, hi = SEQ;
    while (lo < hi) {                      // lower_bound of g in sorted row
        const int mid = (lo + hi) >> 1;
        if (trow[mid] < g) lo = mid + 1; else hi = mid;
    }
    const int rank = off[g * BATCH + row] + (s - lo);
    if (rank < MAXLEN) {
        amap[g * MAXLEN + rank] = inputs[n];
    }
}

// ==================================================================
// Prep kernel 4: W [D x V] f32  ->  Wt [V x D] f16 (transposed convert)
// ==================================================================
__global__ void fl_convw_kernel(const float* __restrict__ W, _Float16* __restrict__ Wt) {
    const int i = blockIdx.x * blockDim.x + threadIdx.x;
    if (i >= DMODEL * NVOCAB) return;
    const int d = i / NVOCAB;
    const int v = i - d * NVOCAB;
    Wt[(long)v * DMODEL + d] = (_Float16)W[i];
}

// ==================================================================
// WMMA GEMM: out[65536 x 1024] = gatherA(f16) @ Wt^T(f16) + bias (f32 acc)
// Block tile 64(M) x 128(N), K-step 32; 8 waves, each 2x2 of 16x16x32.
// Double-buffered LDS, B tile staged via GLOBAL_LOAD_ASYNC_TO_LDS_B128,
// A gather (f32->f16 convert, branchless masked) pipelined across compute.
// ==================================================================
__global__ __launch_bounds__(256) void fl_gemm_kernel(
    const int*      __restrict__ amap,   // [NSLOT] token or -1
    const float*    __restrict__ emb,    // [32000 x 768]
    const _Float16* __restrict__ Wt,     // [1024 x 768] f16
    const float*    __restrict__ bias,   // [1024]
    float*          __restrict__ out)    // [NSLOT x 1024]
{
    __shared__ __attribute__((aligned(64))) _Float16 As[2][64 * 32];    // 2 x 4 KB
    __shared__ __attribute__((aligned(64))) _Float16 Bs[2][128 * 32];   // 2 x 8 KB

    const int tid  = threadIdx.x;
    const int lane = tid & 31;
    const int wave = tid >> 5;
    const int wm   = (wave >> 2) * 32;        // wave M origin: 0 / 32
    const int wn   = (wave & 3) * 32;         // wave N origin: 0..96
    const int nlm  = lane & 15;               // row/col index within 16
    const int khi  = lane >> 4;               // K half-group

    const int blockM = blockIdx.y * 64;
    const int blockN = blockIdx.x * 128;

    // ---- A staging (branchless): thread -> (row 0..63, 8-float chunk 0..3)
    const int ar  = tid >> 2;
    const int aq  = (tid & 3) * 8;
    const int tok = amap[blockM + ar];
    const float  amask = (tok >= 0) ? 1.0f : 0.0f;
    const int    tokc  = (tok >= 0) ? tok : 0;
    const float* asrc  = emb + (long)tokc * DMODEL + aq;   // +kt*32 per step
    const int    adst  = ar * 32 + aq;                     // element offset in As[buf]

    // ---- B staging: 512 16B chunks per K-step; thread owns tid and tid+256.
    // chunk c -> row = c>>2 (0..127), part = c&3 (16B within the 64B row)
    const int brow = tid >> 2;
    const int bprt = (tid & 3) * 8;
    const _Float16* bsrc0 = Wt + (long)(blockN + brow) * DMODEL + bprt;  // +kt*32
    const _Float16* bsrc1 = bsrc0 + 64 * DMODEL;                          // row+64
    const int bdst0 = brow * 32 + bprt;           // element offset in Bs[buf]
    const int bdst1 = bdst0 + 64 * 32;            // +64 rows
#if !FL_ASYNC
    const int brs = tid >> 1;                     // sync fallback staging
    const int bqs = (tid & 1) * 16;
    const _Float16* browf = Wt + (long)(blockN + brs) * DMODEL + bqs;
#endif

    // ---- loop-invariant WMMA operand offsets (elements within a buffer)
    const int arow0 = (wm + nlm) * 32 + khi * 16;         // A contiguous 16 halves
    const int arow1 = arow0 + 16 * 32;                    // +16 M rows
    const int brow0 = (wn + nlm) * 32 + khi * 16;         // B contiguous 16 halves
    const int brow1 = brow0 + 16 * 32;                    // +16 N rows

    v8f acc[2][2] = {};

    // ---- staging helper ----
    auto stage_b = [&](int kt, int buf) {
#if FL_ASYNC
        __builtin_amdgcn_global_load_async_to_lds_b128(
            FL_TO_GLOBAL(bsrc0 + kt * 32), FL_TO_LDS(&Bs[buf][bdst0]), 0, 0);
        __builtin_amdgcn_global_load_async_to_lds_b128(
            FL_TO_GLOBAL(bsrc1 + kt * 32), FL_TO_LDS(&Bs[buf][bdst1]), 0, 0);
#else
        const v8h b0 = *(const v8h*)(browf + kt * 32);
        const v8h b1 = *(const v8h*)(browf + kt * 32 + 8);
        *(v8h*)&Bs[buf][brs * 32 + bqs]     = b0;
        *(v8h*)&Bs[buf][brs * 32 + bqs + 8] = b1;
#endif
    };

    // ---- prologue: stage K-step 0 into buffer 0 ----
    stage_b(0, 0);
    {
        const float4 f0 = *(const float4*)(asrc);
        const float4 f1 = *(const float4*)(asrc + 4);
        *(v8h*)&As[0][adst] = fl_cvt8(f0, f1, amask);
    }
    fl_wait_async();
    __syncthreads();

    // ---- main pipelined loop ----
    for (int kt = 0; kt < KSTEPS; ++kt) {
        const int buf = kt & 1;
        const bool more = (kt + 1) < KSTEPS;

        // issue next-step staging (async B; A global loads into registers)
        float4 f0, f1;
        if (more) {
            stage_b(kt + 1, buf ^ 1);
            f0 = *(const float4*)(asrc + (kt + 1) * 32);
            f1 = *(const float4*)(asrc + (kt + 1) * 32 + 4);
        }

        // ---- compute: 2x2 tiles of v_wmma_f32_16x16x32_f16 ----
        // A-operand (16x32): lane khi holds contiguous K [khi*16, khi*16+16)
        const v8h a0lo = *(const v8h*)&As[buf][arow0];
        const v8h a0hi = *(const v8h*)&As[buf][arow0 + 8];
        const v8h a1lo = *(const v8h*)&As[buf][arow1];
        const v8h a1hi = *(const v8h*)&As[buf][arow1 + 8];
        const v16h a0 = __builtin_shufflevector(a0lo, a0hi, 0,1,2,3,4,5,6,7,8,9,10,11,12,13,14,15);
        const v16h a1 = __builtin_shufflevector(a1lo, a1hi, 0,1,2,3,4,5,6,7,8,9,10,11,12,13,14,15);
        // B-operand (32x16): lane khi holds contiguous K [khi*16, khi*16+16)
        const v16h bb0 = *(const v16h*)&Bs[buf][brow0];
        const v16h bb1 = *(const v16h*)&Bs[buf][brow1];

        acc[0][0] = __builtin_amdgcn_wmma_f32_16x16x32_f16(false, a0, false, bb0,
                                                           (short)0, acc[0][0], false, false);
        acc[0][1] = __builtin_amdgcn_wmma_f32_16x16x32_f16(false, a0, false, bb1,
                                                           (short)0, acc[0][1], false, false);
        acc[1][0] = __builtin_amdgcn_wmma_f32_16x16x32_f16(false, a1, false, bb0,
                                                           (short)0, acc[1][0], false, false);
        acc[1][1] = __builtin_amdgcn_wmma_f32_16x16x32_f16(false, a1, false, bb1,
                                                           (short)0, acc[1][1], false, false);

        if (more) {
            *(v8h*)&As[buf ^ 1][adst] = fl_cvt8(f0, f1, amask);
            fl_wait_async();
            __syncthreads();
        }
    }

    // ---- epilogue: C/D layout: M = vgpr + 8*khi, N = lane&15 ----
    #pragma unroll
    for (int im = 0; im < 2; ++im) {
        #pragma unroll
        for (int in = 0; in < 2; ++in) {
            const int col  = blockN + wn + in * 16 + nlm;
            const float bv = bias[col];
            const long row0 = (long)(blockM + wm + im * 16 + 8 * khi);
            float* op = out + row0 * NVOCAB + col;
            #pragma unroll
            for (int r = 0; r < 8; ++r) {
                op[(long)r * NVOCAB] = acc[im][in][r] + bv;
            }
        }
    }
}

// ==================================================================
extern "C" void kernel_launch(void* const* d_in, const int* in_sizes, int n_in,
                              void* d_out, int out_size, void* d_ws, size_t ws_size,
                              hipStream_t stream) {
    const int*   inputs = (const int*)d_in[0];     // [32*2048] int32
    const int*   target = (const int*)d_in[1];     // [32*2048] int32 (sorted per row)
    const float* emb    = (const float*)d_in[2];   // [32000*768] f32
    const float* W      = (const float*)d_in[3];   // [768*1024] f32
    const float* bias   = (const float*)d_in[4];   // [1024] f32

    // workspace layout
    int*      cnt  = (int*)d_ws;                         // NGROUP*BATCH ints (1 MB)
    int*      amap = cnt + NGROUP * BATCH;               // NSLOT ints (256 KB)
    _Float16* Wt   = (_Float16*)(amap + NSLOT);          // 1024*768 halves (1.5 MB)

    // 0) init scratch
    {
        const int n = NGROUP * BATCH + NSLOT;
        fl_init_kernel<<<(n + 255) / 256, 256, 0, stream>>>(cnt, amap);
    }
    // 1) per-(group,row) counts
    fl_count_kernel<<<NTOK / 256, 256, 0, stream>>>(target, cnt);
    // 2) per-group exclusive row scan
    fl_scan_kernel<<<NGROUP / 256, 256, 0, stream>>>(cnt);
    // 3) ranks -> slot->token map
    fl_rank_kernel<<<NTOK / 256, 256, 0, stream>>>(inputs, target, cnt, amap);
    // 4) W -> Wt (f16, transposed)
    fl_convw_kernel<<<(DMODEL * NVOCAB) / 256, 256, 0, stream>>>(W, Wt);
    // 5) WMMA GEMM + bias
    fl_gemm_kernel<<<dim3(NVOCAB / 128, NSLOT / 64), 256, 0, stream>>>(
        amap, emb, Wt, bias, (float*)d_out);
}
